// HoLSTM_31293131718759
// MI455X (gfx1250) — matrix operand order repaired and checked by reading, not compiled, not run on hardware
//
#include <hip/hip_runtime.h>
#include <hip/hip_bf16.h>
#include <stdint.h>

#define SEQ 512
#define BB  2048
#define HH  512
#define XF  31   // x feature dim (IN - OUT)

typedef __attribute__((ext_vector_type(16))) __bf16        v16bf;
typedef __attribute__((ext_vector_type(8)))  float         v8f;
typedef __attribute__((ext_vector_type(16))) unsigned short v16us;

__device__ __forceinline__ unsigned short f2bf(float f) {
  unsigned int u = __float_as_uint(f);
  u += 0x7FFFu + ((u >> 16) & 1u);          // round-to-nearest-even
  return (unsigned short)(u >> 16);
}
__device__ __forceinline__ float bf2f(unsigned short h) {
  return __uint_as_float(((unsigned int)h) << 16);
}
__device__ __forceinline__ float sigm(float x) {
  return 1.0f / (1.0f + __expf(-x));
}
__device__ __forceinline__ v8f wmma_bf16(v16bf a, v16bf b, v8f c) {
  // D = A(16x32 bf16) * B(32x16 bf16) + C(16x16 f32)
  return __builtin_amdgcn_wmma_f32_16x16x32_bf16(false, a, false, b, (short)0, c,
                                                 false, false);
}

// ---- A-fragment swizzle: matrices consumed as WMMA A (x0, h) are stored so
// that each lane's 16 fragment values are contiguous (one 32B load, no moves).
// ISA A 16x32 bf16 layout: lanes 0-15 (row M=lane) hold K {0..7,16..23};
// lanes 16-31 hold K {8..15,24..31}.
__device__ __forceinline__ size_t swzA(int row, int k) {
  size_t blk = ((size_t)(row >> 4) * 16 + (k >> 5)) * 32 + ((k >> 3) & 1) * 16 +
               (row & 15);
  return blk * 16 + (k & 7) + (((k >> 4) & 1) << 3);
}
// Load one A fragment from a swizzled buffer: tile row base (mult of 16), k-tile kt.
__device__ __forceinline__ v16bf loadA_swz(const unsigned short* base, int row0,
                                           int kt, int lane) {
  const unsigned short* p =
      base + (((size_t)(row0 >> 4) * 16 + kt) * 32 + lane) * 16;
  v16us r = *reinterpret_cast<const v16us*>(p);
  return __builtin_bit_cast(v16bf, r);
}

// B-fragment: B = W^T where W is row-major [N, K].
// ISA layout: lanes 0-15 col n hold K {0..7,16..23}; lanes 16-31 K {8..15,24..31}.
__device__ __forceinline__ v16bf loadB(const unsigned short* W, int n0,
                                       int stride, int kk, int lane) {
  typedef __attribute__((ext_vector_type(8))) unsigned short v8us;
  int n = n0 + (lane & 15);
  const unsigned short* p = W + (size_t)n * stride + kk + ((lane & 16) ? 8 : 0);
  v8us lo = *reinterpret_cast<const v8us*>(p);
  v8us hi = *reinterpret_cast<const v8us*>(p + 16);
  v16us r = __builtin_shufflevector(lo, hi, 0, 1, 2, 3, 4, 5, 6, 7, 8, 9, 10, 11, 12, 13, 14, 15);
  return __builtin_bit_cast(v16bf, r);
}

// ---------------- prep kernels ----------------
__global__ void cvt_bf16_kernel(const float* __restrict__ src,
                                unsigned short* __restrict__ dst, int n) {
  int i = blockIdx.x * blockDim.x + threadIdx.x;
  if (i < n) dst[i] = f2bf(src[i]);
}
// convert [rows x 512] f32 into the swizzled A layout
__global__ void cvt_bf16_swz_kernel(const float* __restrict__ src,
                                    unsigned short* __restrict__ dst, int n) {
  int i = blockIdx.x * blockDim.x + threadIdx.x;
  if (i < n) dst[swzA(i >> 9, i & 511)] = f2bf(src[i]);
}
// W_out [512] -> swizzled as row 0 of a 16-row tile (to pair with swizzled h)
__global__ void cvt_wout_swz_kernel(const float* __restrict__ src,
                                    unsigned short* __restrict__ dst) {
  int k = blockIdx.x * blockDim.x + threadIdx.x;
  if (k < HH) dst[swzA(0, k)] = f2bf(src[k]);
}
__global__ void copy_f32_kernel(const float* __restrict__ src,
                                float* __restrict__ dst, int n) {
  int i = blockIdx.x * blockDim.x + threadIdx.x;
  if (i < n) dst[i] = src[i];
}
__global__ void bias_sum_kernel(const float* __restrict__ a,
                                const float* __restrict__ b,
                                float* __restrict__ dst, int n) {
  int i = blockIdx.x * blockDim.x + threadIdx.x;
  if (i < n) dst[i] = a[i] + b[i];
}

// ---------------- per-step kernel 0: x0 = relu([x_t|y] @ W_in^T + b_in) ----------------
// 4096 16x16 tiles, one wave per tile, 8 waves/block -> 512 blocks. Output swizzled.
__global__ void x0_kernel(const float* __restrict__ xt,      // [B, 31]
                          const float* __restrict__ ybuf,    // [B]
                          const unsigned short* __restrict__ Winb, // [H, 32] bf16
                          const float* __restrict__ b_in,    // [H]
                          unsigned short* __restrict__ x0) { // [B, H] bf16, swizzled
  int wave = threadIdx.x >> 5;
  int lane = threadIdx.x & 31;
  int tile = blockIdx.x * 8 + wave;      // 0..4095
  int mt = tile >> 5;                    // 0..127  (batch tile)
  int nt = tile & 31;                    // 0..31   (hidden tile)

  // Build A fragment (inp = [x | y], K = 32), per ISA 16x32 bf16 layout.
  int row = mt * 16 + (lane & 15);
  int kb  = (lane & 16) ? 8 : 0;
  float yv = ybuf[row];
  v16us au;
#pragma unroll
  for (int i = 0; i < 8; ++i) {
    int k1 = kb + 16 + i;   // 16..31 -> col 31 is y
    float f0 = xt[(size_t)row * XF + kb + i];          // 0..15: always from x
    float f1 = (k1 < XF) ? xt[(size_t)row * XF + k1] : yv;
    au[i]     = f2bf(f0);
    au[i + 8] = f2bf(f1);
  }
  v16bf a = __builtin_bit_cast(v16bf, au);
  v16bf b = loadB(Winb, nt * 16, 32, 0, lane);

  v8f acc = {};
  acc = wmma_bf16(a, b, acc);

  // epilogue: bias + relu + bf16, scatter into swizzled x0
  int col  = nt * 16 + (lane & 15);
  float bias = b_in[col];
  size_t cbase = ((size_t)(mt * 16 + (col >> 5)) * 32 + ((col >> 3) & 1) * 16) * 16 +
                 (col & 7) + (((col >> 4) & 1) << 3);
#pragma unroll
  for (int j = 0; j < 8; ++j) {
    int rlow = (lane & 16) ? (j + 8) : j;
    float v = acc[j] + bias;
    v = v > 0.0f ? v : 0.0f;
    x0[cbase + (size_t)rlow * 16] = f2bf(v);
  }
}

// ---------------- per-step kernel 1: gates GEMM + LSTM pointwise ----------------
// block = 64 batch rows x 64 hidden cols (x4 gates); 8 waves of 32x16.
__global__ void gates_kernel(const unsigned short* __restrict__ x0,    // [B,H] bf16 swz
                             const unsigned short* __restrict__ hprev, // [B,H] bf16 swz
                             const unsigned short* __restrict__ Wih,   // [4H,H] bf16
                             const unsigned short* __restrict__ Whh,   // [4H,H] bf16
                             const float* __restrict__ bgate,          // [4H]
                             float* __restrict__ c,                    // [B,H] f32 (in place)
                             unsigned short* __restrict__ hnext) {     // [B,H] bf16 swz
  int wave = threadIdx.x >> 5;
  int lane = threadIdx.x & 31;
  int wm = wave & 1;        // M half
  int wn = wave >> 1;       // N quarter
  int rb = blockIdx.x * 64 + wm * 32;          // batch row base (32 rows)
  int n0 = blockIdx.y * 64 + wn * 16;          // hidden col tile base

  v8f acc[4][2];
#pragma unroll
  for (int g = 0; g < 4; ++g)
#pragma unroll
    for (int m = 0; m < 2; ++m) acc[g][m] = (v8f){};

  for (int kt = 0; kt < HH / 32; ++kt) {
    int kk = kt * 32;
    v16bf aX[2], aH[2];
#pragma unroll
    for (int m = 0; m < 2; ++m) {
      aX[m] = loadA_swz(x0,    rb + m * 16, kt, lane);
      aH[m] = loadA_swz(hprev, rb + m * 16, kt, lane);
    }
#pragma unroll
    for (int g = 0; g < 4; ++g) {
      v16bf bi = loadB(Wih, g * HH + n0, HH, kk, lane);
      v16bf bh = loadB(Whh, g * HH + n0, HH, kk, lane);
#pragma unroll
      for (int m = 0; m < 2; ++m) {
        acc[g][m] = wmma_bf16(aX[m], bi, acc[g][m]);
        acc[g][m] = wmma_bf16(aH[m], bh, acc[g][m]);
      }
    }
  }

  int col = n0 + (lane & 15);
  float bi_ = bgate[0 * HH + col];
  float bf_ = bgate[1 * HH + col];
  float bg_ = bgate[2 * HH + col];
  float bo_ = bgate[3 * HH + col];
  // swizzled-store components for hnext (col part fixed per lane)
  size_t ce = (size_t)((col >> 3) & 1) * 16 * 16 + (col & 7) + (((col >> 4) & 1) << 3);
  int    ckt = col >> 5;

#pragma unroll
  for (int m = 0; m < 2; ++m) {
    int mt = (rb >> 4) + m;
#pragma unroll
    for (int j = 0; j < 8; ++j) {
      int rlow = (lane & 16) ? (j + 8) : j;
      int row = mt * 16 + rlow;
      size_t idx = (size_t)row * HH + col;          // linear (for c)
      float iv = sigm(acc[0][m][j] + bi_);
      float fv = sigm(acc[1][m][j] + bf_);
      float gv = tanhf(acc[2][m][j] + bg_);
      float ov = sigm(acc[3][m][j] + bo_);
      float cn = fv * c[idx] + iv * gv;
      c[idx] = cn;
      size_t sidx = ((size_t)(mt * 16 + ckt) * 32) * 16 + ce + (size_t)rlow * 16;
      hnext[sidx] = f2bf(ov * tanhf(cn));
    }
  }
}

// ---------------- per-step kernel 2: y = h @ W_out^T + b_out ----------------
// h and W_out both in the A-swizzle; lane l owns block (kt=l&15, lanehi=l>>4).
__global__ void y_kernel(const unsigned short* __restrict__ h,     // [B,H] bf16 swz
                         const unsigned short* __restrict__ Woutb, // swizzled [16*32*16]
                         const float* __restrict__ b_out,          // [1]
                         float* __restrict__ ybuf,                 // [B]
                         float* __restrict__ out_t) {              // [B]
  int lane = threadIdx.x & 31;
  int row  = blockIdx.x * 8 + (threadIdx.x >> 5);
  int kt = lane & 15, lhi = lane >> 4;
  const unsigned short* hp =
      h + (((size_t)(row >> 4) * 16 + kt) * 32 + lhi * 16 + (row & 15)) * 16;
  const unsigned short* wp = Woutb + ((size_t)kt * 32 + lhi * 16) * 16;
  v16us hv = *reinterpret_cast<const v16us*>(hp);
  v16us wv = *reinterpret_cast<const v16us*>(wp);
  float p = 0.0f;
#pragma unroll
  for (int e = 0; e < 16; ++e) p += bf2f(hv[e]) * bf2f(wv[e]);
#pragma unroll
  for (int off = 16; off >= 1; off >>= 1) p += __shfl_xor(p, off, 32);
  if (lane == 0) {
    float y = p + b_out[0];
    ybuf[row]  = y;
    out_t[row] = y;
  }
}

// ---------------- launch ----------------
static inline size_t alignup(size_t x) { return (x + 255) & ~(size_t)255; }

extern "C" void kernel_launch(void* const* d_in, const int* in_sizes, int n_in,
                              void* d_out, int out_size, void* d_ws, size_t ws_size,
                              hipStream_t stream) {
  (void)in_sizes; (void)n_in; (void)out_size; (void)ws_size;
  const float* x     = (const float*)d_in[0];   // [SEQ,B,31]
  const float* h0    = (const float*)d_in[1];   // [B,H]
  const float* c0    = (const float*)d_in[2];   // [B,H]
  const float* y0    = (const float*)d_in[3];   // [B,1]
  const float* W_in  = (const float*)d_in[4];   // [H,32]
  const float* b_in  = (const float*)d_in[5];   // [H]
  const float* W_ih  = (const float*)d_in[6];   // [4H,H]
  const float* W_hh  = (const float*)d_in[7];   // [4H,H]
  const float* b_ih  = (const float*)d_in[8];   // [4H]
  const float* b_hh  = (const float*)d_in[9];   // [4H]
  const float* W_out = (const float*)d_in[10];  // [1,H]
  const float* b_out = (const float*)d_in[11];  // [1]
  float* out = (float*)d_out;                   // [SEQ,B]

  char* ws = (char*)d_ws;
  size_t off = 0;
  auto take = [&](size_t bytes) { size_t o = off; off = alignup(off + bytes); return o; };
  unsigned short* WinB  = (unsigned short*)(ws + take((size_t)HH * 32 * 2));
  unsigned short* WihB  = (unsigned short*)(ws + take((size_t)4 * HH * HH * 2));
  unsigned short* WhhB  = (unsigned short*)(ws + take((size_t)4 * HH * HH * 2));
  unsigned short* WoutB = (unsigned short*)(ws + take((size_t)16 * 32 * 16 * 2));
  float*          bgate = (float*)(ws + take((size_t)4 * HH * 4));
  unsigned short* hA    = (unsigned short*)(ws + take((size_t)BB * HH * 2));
  unsigned short* hB2   = (unsigned short*)(ws + take((size_t)BB * HH * 2));
  float*          cbuf  = (float*)(ws + take((size_t)BB * HH * 4));
  float*          ybuf  = (float*)(ws + take((size_t)BB * 4));
  unsigned short* x0buf = (unsigned short*)(ws + take((size_t)BB * HH * 2));

  auto grid1 = [](int n) { return dim3((n + 255) / 256); };

  // Prep: bf16 weight/state conversion, bias fusion, state init (re-done every call).
  cvt_bf16_kernel<<<grid1(HH * 32), 256, 0, stream>>>(W_in, WinB, HH * 32);
  cvt_bf16_kernel<<<grid1(4 * HH * HH), 256, 0, stream>>>(W_ih, WihB, 4 * HH * HH);
  cvt_bf16_kernel<<<grid1(4 * HH * HH), 256, 0, stream>>>(W_hh, WhhB, 4 * HH * HH);
  cvt_wout_swz_kernel<<<grid1(HH), 256, 0, stream>>>(W_out, WoutB);
  cvt_bf16_swz_kernel<<<grid1(BB * HH), 256, 0, stream>>>(h0, hA, BB * HH);
  bias_sum_kernel<<<grid1(4 * HH), 256, 0, stream>>>(b_ih, b_hh, bgate, 4 * HH);
  copy_f32_kernel<<<grid1(BB * HH), 256, 0, stream>>>(c0, cbuf, BB * HH);
  copy_f32_kernel<<<grid1(BB), 256, 0, stream>>>(y0, ybuf, BB);

  unsigned short* hp[2] = { hA, hB2 };
  for (int t = 0; t < SEQ; ++t) {
    const float* xt = x + (size_t)t * BB * XF;
    unsigned short* hprev = hp[t & 1];
    unsigned short* hnext = hp[(t + 1) & 1];
    x0_kernel<<<512, 256, 0, stream>>>(xt, ybuf, WinB, b_in, x0buf);
    gates_kernel<<<dim3(BB / 64, HH / 64), 256, 0, stream>>>(x0buf, hprev, WihB, WhhB,
                                                             bgate, cbuf, hnext);
    y_kernel<<<BB / 8, 256, 0, stream>>>(hnext, WoutB, b_out, ybuf, out + (size_t)t * BB);
  }
}
